// CGCN_55482387529958
// MI455X (gfx1250) — matrix operand order repaired and checked by reading, not compiled, not run on hardware
//
#include <hip/hip_runtime.h>
#include <math.h>

// ---------------- problem constants (from reference) ----------------
#define NNODES 50000
#define NFEAT  500
#define NHID   128
#define NCLS   16
#define NREL   3
#define NEDGE  300000
#define KPAD   512            // 500 padded to 16 chunks of K=32
#define ALPHA_ 0.1f
#define LAMDA2_ 1.0f
#define LAMDA1_ 9.0f          // 1/alpha - 1
#define THRESH_C_ 1e-6f
#define NUM_ITERS_ 50

typedef __attribute__((ext_vector_type(16))) __bf16         v16bf;
typedef __attribute__((ext_vector_type(16))) unsigned short v16u;
typedef __attribute__((ext_vector_type(8)))  float          v8f;
typedef __attribute__((ext_vector_type(4)))  unsigned int   u32x4;
typedef __attribute__((ext_vector_type(8)))  int            i32x8;
typedef __attribute__((ext_vector_type(4)))  int            i32x4;

#if __has_builtin(__builtin_amdgcn_tensor_load_to_lds) && __has_builtin(__builtin_amdgcn_s_wait_tensorcnt)
#define USE_TDM 1
#else
#define USE_TDM 0
#endif

__device__ __forceinline__ unsigned short f32_to_bf16(float f) {
    unsigned int u = __builtin_bit_cast(unsigned int, f);
    unsigned int r = u + 0x7FFFu + ((u >> 16) & 1u);   // round-to-nearest-even
    return (unsigned short)(r >> 16);
}

// ---------------- small utility kernels ----------------
__global__ void fill_f32(float* p, int n, float v) {
    int i = blockIdx.x * blockDim.x + threadIdx.x;
    if (i < n) p[i] = v;
}

// W1 [128,500] f32 row-major -> W1T bf16 [512][128] (zero-padded K)
__global__ void conv_w1t(const float* __restrict__ w1, unsigned short* __restrict__ w1t) {
    int i = blockIdx.x * blockDim.x + threadIdx.x;
    if (i >= KPAD * NHID) return;
    int k = i / NHID, n = i - k * NHID;
    float v = (k < NFEAT) ? w1[n * NFEAT + k] : 0.0f;
    w1t[i] = f32_to_bf16(v);
}

// W2 [16,128] f32 -> W2T bf16 [128][16]
__global__ void conv_w2t(const float* __restrict__ w2, unsigned short* __restrict__ w2t) {
    int i = blockIdx.x * blockDim.x + threadIdx.x;
    if (i >= NHID * NCLS) return;
    int k = i / NCLS, n = i - k * NCLS;
    w2t[i] = f32_to_bf16(w2[n * NHID + k]);
}

// ---------------- GEMM1: h = relu(x @ W1^T + b1), also raw = h ----------------
// block = 128 threads (4 waves); block computes a 16x128 output tile.
// K is processed in 4 groups of 128; each group's W1T panel (128 K x 128 N bf16,
// 32 KB) is DMA'd into LDS by the Tensor Data Mover (wave 0), then all waves
// consume it through ds_load while the x-tile (HBM-bound stream) stays on the
// vector-memory path.
__global__ __launch_bounds__(128)
void gemm1_wmma(const float* __restrict__ x, const unsigned short* __restrict__ w1t,
                const float* __restrict__ b1, float* __restrict__ h, float* __restrict__ raw) {
    __shared__ unsigned short sA[16 * KPAD];    // 16 KB: A tile (bf16)
#if USE_TDM
    __shared__ unsigned short sB[128 * NHID];   // 32 KB: one K-group of W1T
#endif
    const int mtile = blockIdx.x;               // 0..3124 (3125*16 == 50000 exactly)
    const int tid   = threadIdx.x;

    // Stage A tile: 16 rows x 500 f32 -> bf16 LDS, zero-pad to 512
    for (int i = tid; i < 16 * KPAD; i += 128) {
        int r = i >> 9, k = i & (KPAD - 1);
        float v = (k < NFEAT) ? x[(mtile * 16 + r) * NFEAT + k] : 0.0f;
        sA[i] = f32_to_bf16(v);
    }

    const int wave = tid >> 5, lane = tid & 31;
    const int n0 = wave * 32;                   // base column of this wave's 2 tiles
    const int mA = lane & 15;
    const int kb = (lane < 16) ? 0 : 8;         // ISA 16-bit A layout half-wave K base
    v8f c0 = {}; v8f c1 = {};

    for (int kg = 0; kg < KPAD / 128; ++kg) {
        __syncthreads();                        // prior sB reads done; A staged (kg==0)
#if USE_TDM
        if (wave == 0) {
            // Tensor DMA descriptor (ISA ch.8): 2-D tile, data_size=4B,
            // row = 64 dwords (128 bf16), 128 rows, contiguous (stride 64).
            unsigned long long ga =
                (unsigned long long)(const void*)(w1t + (size_t)kg * 128 * NHID);
            unsigned int lds = (unsigned int)(unsigned long long)(void*)&sB[0];
            u32x4 g0;
            g0[0] = 1u;                                              // count=1
            g0[1] = lds;                                             // lds_addr
            g0[2] = (unsigned int)ga;                                // global_addr lo
            g0[3] = (unsigned int)((ga >> 32) & 0x01FFFFFFu) | 0x80000000u; // hi | type=2
            i32x8 g1;
            g1[0] = 0x00020000;         // workgroup_mask=0, data_size=2 (4 bytes)
            g1[1] = (int)(64u << 16);   // tensor_dim0 = 64 (lo16 @ bits 63:48)
            g1[2] = (int)(128u << 16);  // tensor_dim1 = 128 (lo16 @ bits 95:80)
            g1[3] = (int)(64u << 16);   // tile_dim0 = 64  (bits 127:112)
            g1[4] = 128;                // tile_dim1 = 128 (bits 143:128)
            g1[5] = 64;                 // tensor_dim0_stride = 64 dwords
            g1[6] = 0; g1[7] = 0;
            i32x4 g2 = {0, 0, 0, 0}, g3 = {0, 0, 0, 0};
#if defined(__clang_major__) && (__clang_major__ >= 23)
            i32x8 gpad = {0, 0, 0, 0, 0, 0, 0, 0};
            __builtin_amdgcn_tensor_load_to_lds(g0, g1, g2, g3, gpad, 0);
#else
            __builtin_amdgcn_tensor_load_to_lds(g0, g1, g2, g3, 0);
#endif
            __builtin_amdgcn_s_wait_tensorcnt(0);
        }
#endif
        __syncthreads();                        // sB ready for all waves

        #pragma unroll
        for (int j = 0; j < 4; ++j) {
            const int kc = kg * 4 + j;
            // A fragment: lane holds M=mA; halves 0..7 -> K=kb.., 8..15 -> K=16+kb..
            const unsigned short* pa = &sA[mA * KPAD + kc * 32];
            v16u au;
            #pragma unroll
            for (int q = 0; q < 8; ++q) { au[q] = pa[kb + q]; au[8 + q] = pa[16 + kb + q]; }
            v16bf a = __builtin_bit_cast(v16bf, au);
            // B fragments: lane holds N=lane%16; halves 0..7 -> K=kb.., 8..15 -> K=16+kb..
#if USE_TDM
            const unsigned short* pb = &sB[(j * 32) * NHID + mA];
#else
            const unsigned short* pb = &w1t[(kc * 32) * NHID + mA];
#endif
            v16u b0u, b1u;
            #pragma unroll
            for (int q = 0; q < 8; ++q) {
                b0u[q] = pb[(kb + q) * NHID + n0]; b0u[8 + q] = pb[(16 + kb + q) * NHID + n0];
                b1u[q] = pb[(kb + q) * NHID + n0 + 16]; b1u[8 + q] = pb[(16 + kb + q) * NHID + n0 + 16];
            }
            v16bf b0 = __builtin_bit_cast(v16bf, b0u);
            v16bf b1 = __builtin_bit_cast(v16bf, b1u);
            c0 = __builtin_amdgcn_wmma_f32_16x16x32_bf16(false, a, false, b0, (short)0, c0, false, false);
            c1 = __builtin_amdgcn_wmma_f32_16x16x32_bf16(false, a, false, b1, (short)0, c1, false, false);
        }
    }

    // D layout: VGPR g, lane L -> M = g + 8*(L/16), N = L%16
    const int nlane = lane & 15, mofs = (lane >> 4) * 8, mrow = mtile * 16;
    #pragma unroll
    for (int g = 0; g < 8; ++g) {
        int m = mrow + mofs + g;
        int nA = n0 + nlane, nB = n0 + 16 + nlane;
        float vA = c0[g] + b1[nA]; vA = vA > 0.f ? vA : 0.f;
        float vB = c1[g] + b1[nB]; vB = vB > 0.f ? vB : 0.f;
        h[m * NHID + nA] = vA;  raw[m * NHID + nA] = vA;
        h[m * NHID + nB] = vB;  raw[m * NHID + nB] = vB;
    }
}

// ---------------- degree kernels (edge structure only; computed once) ----------------
__global__ void deg_kernel(const int* __restrict__ ei, float* __restrict__ deg) {
    int i = blockIdx.x * blockDim.x + threadIdx.x;
    if (i >= NREL * NEDGE) return;
    int r = i / NEDGE, e = i - r * NEDGE;
    int row = ei[(r * 2 + 0) * NEDGE + e];
    atomicAdd(&deg[r * NNODES + row], 1.0f);
}

__global__ void degfin_kernel(const float* __restrict__ deg,
                              float* __restrict__ dm05, float* __restrict__ dm025) {
    int i = blockIdx.x * blockDim.x + threadIdx.x;
    if (i >= NREL * NNODES) return;
    float d = deg[i]; d = d > 1.0f ? d : 1.0f;
    float r05 = rsqrtf(d);          // d^-0.5
    dm05[i]  = r05;
    dm025[i] = sqrtf(r05);          // d^-0.25
}

// ---------------- per-edge distance + s_r accumulation (one wave per edge) ----------------
__global__ __launch_bounds__(256)
void dist_kernel(const int* __restrict__ ei, const float* __restrict__ h,
                 const float* __restrict__ dm025, float* __restrict__ dist2,
                 float* __restrict__ s_sum) {
    int wid  = blockIdx.x * 8 + (threadIdx.x >> 5);
    int lane = threadIdx.x & 31;
    if (wid >= NREL * NEDGE) return;
    int r = wid / NEDGE, e = wid - r * NEDGE;
    int row = ei[(r * 2 + 0) * NEDGE + e];
    int col = ei[(r * 2 + 1) * NEDGE + e];
    float sa = dm025[r * NNODES + row];
    float sb = dm025[r * NNODES + col];
    float4 va = ((const float4*)&h[row * NHID])[lane];
    float4 vb = ((const float4*)&h[col * NHID])[lane];
    float dx = va.x * sa - vb.x * sb, dy = va.y * sa - vb.y * sb;
    float dz = va.z * sa - vb.z * sb, dw = va.w * sa - vb.w * sb;
    float acc = dx * dx + dy * dy + dz * dz + dw * dw;
    #pragma unroll
    for (int m = 16; m >= 1; m >>= 1) acc += __shfl_xor(acc, m, 32);
    if (lane == 0) {
        float d2 = acc + 1e-4f;
        dist2[r * NEDGE + e] = d2;
        atomicAdd(&s_sum[r], sqrtf(d2));     // dist2^(gamma/2), gamma=1
    }
}

// ---------------- mirror-descent coefficient loop (tiny; replicates fori_loop) ----------------
__global__ void u_kernel(const float* __restrict__ s_sum, float* __restrict__ u_out) {
    if (threadIdx.x != 0 || blockIdx.x != 0) return;
    float s[NREL], l1 = 0.f;
    for (int r = 0; r < NREL; ++r) { s[r] = s_sum[r] / (float)NEDGE; l1 += fabsf(s[r]); }
    float fi = l1 + 2.0f * LAMDA2_ / LAMDA1_;
    float u[NREL]; for (int r = 0; r < NREL; ++r) u[r] = 1.0f / (float)NREL;
    bool done = false;
    for (int i = 0; i < NUM_ITERS_; ++i) {
        float t = (float)i + 1.0f;
        float Tt = sqrtf(2.0f * logf((float)NREL) / (t * fi * fi));
        float ut[NREL], ss = 0.f;
        for (int r = 0; r < NREL; ++r) {
            float fde = 2.0f * LAMDA2_ / LAMDA1_ * u[r] + s[r];
            ut[r] = u[r] * expf(-Tt * fde); ss += ut[r];
        }
        float cond = 0.f, un[NREL];
        for (int r = 0; r < NREL; ++r) {
            float nu = done ? u[r] : (ut[r] / ss);
            un[r] = nu; cond += (u[r] - nu) * (u[r] - nu);
        }
        for (int r = 0; r < NREL; ++r) u[r] = un[r];
        done = done || (sqrtf(cond) < THRESH_C_);
    }
    for (int r = 0; r < NREL; ++r) u_out[r] = u[r];
}

// ---------------- hout = alpha*raw ----------------
__global__ void init_hout(const float* __restrict__ raw, float* __restrict__ hout, int n) {
    int i = blockIdx.x * blockDim.x + threadIdx.x;
    if (i < n) hout[i] = ALPHA_ * raw[i];
}

// ---------------- message passing scatter (one wave per edge) ----------------
__global__ __launch_bounds__(256)
void mp_kernel(const int* __restrict__ ei, const float* __restrict__ h,
               const float* __restrict__ dm05, const float* __restrict__ dist2,
               const float* __restrict__ u, float* __restrict__ hout) {
    int wid  = blockIdx.x * 8 + (threadIdx.x >> 5);
    int lane = threadIdx.x & 31;
    if (wid >= NREL * NEDGE) return;
    int r = wid / NEDGE, e = wid - r * NEDGE;
    int row = ei[(r * 2 + 0) * NEDGE + e];
    int col = ei[(r * 2 + 1) * NEDGE + e];
    float d2   = dist2[r * NEDGE + e];
    float g    = tanhf(rsqrtf(d2));                       // tanh(dist2^{-gamma/2})
    float coef = (1.0f - ALPHA_) * u[r] * g *
                 dm05[r * NNODES + row] * dm05[r * NNODES + col];
    float4 va = ((const float4*)&h[row * NHID])[lane];
    float* po = &hout[col * NHID + lane * 4];
    atomicAdd(po + 0, coef * va.x);
    atomicAdd(po + 1, coef * va.y);
    atomicAdd(po + 2, coef * va.z);
    atomicAdd(po + 3, coef * va.w);
}

// ---------------- GEMM2 + fused log-softmax (one wave per 16-row tile) ----------------
__global__ __launch_bounds__(256)
void gemm2_wmma(const float* __restrict__ h, const unsigned short* __restrict__ w2t,
                const float* __restrict__ b2, float* __restrict__ out) {
    int wave = threadIdx.x >> 5, lane = threadIdx.x & 31;
    int mtile = blockIdx.x * 8 + wave;
    if (mtile >= NNODES / 16) return;
    const int mA = lane & 15, kb = (lane < 16) ? 0 : 8;
    v8f c = {};
    for (int kc = 0; kc < NHID / 32; ++kc) {
        const float* pa = &h[(mtile * 16 + mA) * NHID + kc * 32];
        v16u au;
        #pragma unroll
        for (int j = 0; j < 8; ++j) {
            au[j]     = f32_to_bf16(pa[kb + j]);
            au[8 + j] = f32_to_bf16(pa[16 + kb + j]);
        }
        v16bf a = __builtin_bit_cast(v16bf, au);
        const unsigned short* pb = &w2t[(kc * 32) * NCLS + mA];
        v16u bu;
        #pragma unroll
        for (int j = 0; j < 8; ++j) { bu[j] = pb[(kb + j) * NCLS]; bu[8 + j] = pb[(16 + kb + j) * NCLS]; }
        v16bf b = __builtin_bit_cast(v16bf, bu);
        c = __builtin_amdgcn_wmma_f32_16x16x32_bf16(false, a, false, b, (short)0, c, false, false);
    }
    // Row M of D lives in one VGPR across a 16-lane half-wave -> xor 1,2,4,8 stays in-row
    const int nlane = lane & 15, mofs = (lane >> 4) * 8;
    #pragma unroll
    for (int g = 0; g < 8; ++g) {
        float v = c[g] + b2[nlane];
        float mx = v;
        #pragma unroll
        for (int m = 1; m <= 8; m <<= 1) mx = fmaxf(mx, __shfl_xor(mx, m, 32));
        float ex = expf(v - mx);
        float sm = ex;
        #pragma unroll
        for (int m = 1; m <= 8; m <<= 1) sm += __shfl_xor(sm, m, 32);
        float lsm = (v - mx) - logf(sm);
        int mrow = mtile * 16 + mofs + g;
        out[mrow * NCLS + nlane] = lsm;                         // log_softmax
        out[NNODES * NCLS + mrow * NCLS + nlane] = v;           // logits
    }
}

// ---------------- host orchestration ----------------
extern "C" void kernel_launch(void* const* d_in, const int* in_sizes, int n_in,
                              void* d_out, int out_size, void* d_ws, size_t ws_size,
                              hipStream_t stream) {
    const float* x  = (const float*)d_in[0];
    const int*   ei = (const int*)  d_in[1];
    const float* W1 = (const float*)d_in[2];
    const float* b1 = (const float*)d_in[3];
    const float* W2 = (const float*)d_in[4];
    const float* b2 = (const float*)d_in[5];
    float* out = (float*)d_out;

    // carve workspace (256B aligned)
    char* base = (char*)d_ws;
    size_t off = 0;
    auto carve = [&](size_t bytes) { char* p = base + off; off = (off + bytes + 255) & ~(size_t)255; return p; };
    unsigned short* w1t  = (unsigned short*)carve((size_t)KPAD * NHID * 2);
    unsigned short* w2t  = (unsigned short*)carve((size_t)NHID * NCLS * 2);
    float* hA    = (float*)carve((size_t)NNODES * NHID * 4);
    float* hB    = (float*)carve((size_t)NNODES * NHID * 4);
    float* raw   = (float*)carve((size_t)NNODES * NHID * 4);
    float* deg   = (float*)carve((size_t)NREL * NNODES * 4);
    float* dm05  = (float*)carve((size_t)NREL * NNODES * 4);
    float* dm025 = (float*)carve((size_t)NREL * NNODES * 4);
    float* dist2 = (float*)carve((size_t)NREL * NEDGE * 4);
    float* s_sum = (float*)carve(64);
    float* u     = (float*)carve(64);
    (void)ws_size; (void)in_sizes; (void)n_in; (void)out_size;

    // weight conversion
    conv_w1t<<<(KPAD * NHID + 255) / 256, 256, 0, stream>>>(W1, w1t);
    conv_w2t<<<(NHID * NCLS + 255) / 256, 256, 0, stream>>>(W2, w2t);

    // degrees (edge structure only)
    fill_f32<<<(NREL * NNODES + 255) / 256, 256, 0, stream>>>(deg, NREL * NNODES, 0.0f);
    deg_kernel<<<(NREL * NEDGE + 255) / 256, 256, 0, stream>>>(ei, deg);
    degfin_kernel<<<(NREL * NNODES + 255) / 256, 256, 0, stream>>>(deg, dm05, dm025);

    // h = relu(x @ W1^T + b1); raw = h
    gemm1_wmma<<<NNODES / 16, 128, 0, stream>>>(x, w1t, b1, hA, raw);

    const int edge_blocks = (NREL * NEDGE + 7) / 8;       // 1 wave per edge, 8 waves/block
    float* hcur = hA; float* hnxt = hB;
    for (int layer = 0; layer < 2; ++layer) {
        fill_f32<<<1, 32, 0, stream>>>(s_sum, NREL, 0.0f);
        dist_kernel<<<edge_blocks, 256, 0, stream>>>(ei, hcur, dm025, dist2, s_sum);
        u_kernel<<<1, 32, 0, stream>>>(s_sum, u);
        init_hout<<<(NNODES * NHID + 255) / 256, 256, 0, stream>>>(raw, hnxt, NNODES * NHID);
        mp_kernel<<<edge_blocks, 256, 0, stream>>>(ei, hcur, dm05, dist2, u, hnxt);
        float* t = hcur; hcur = hnxt; hnxt = t;
    }

    // logits + log_softmax -> d_out (log_softmax first, then logits)
    gemm2_wmma<<<(NNODES / 16 + 7) / 8, 256, 0, stream>>>(hcur, w2t, b2, out);
}
